// TransformerDecoder_214748364917
// MI455X (gfx1250) — compile-verified, matrix operand order checked by reading, not run on hardware
//
#include <hip/hip_runtime.h>
#include <math.h>

typedef unsigned short bf16_t;
typedef __attribute__((ext_vector_type(16))) __bf16 v16bf;
typedef __attribute__((ext_vector_type(8)))  float  v8f;

#define KP 40   // LDS row pitch in bf16 elements (80B, 16B-aligned, conflict-spread)

__device__ __forceinline__ unsigned short f2bf(float f) {
  unsigned int u = __float_as_uint(f);
  unsigned int r = u + 0x7FFFu + ((u >> 16) & 1u);
  return (unsigned short)(r >> 16);
}

union FragU { uint4 u[2]; v16bf v; };

// 32 bytes global -> LDS, per lane, via CDNA5 async copy (ASYNCcnt-tracked).
// INST_OFFSET applies to both the LDS and the global address.
__device__ __forceinline__ void async_cp32(unsigned ldsOff, const bf16_t* g) {
  asm volatile("global_load_async_to_lds_b128 %0, %1, off"
               :: "v"(ldsOff), "v"(g) : "memory");
  asm volatile("global_load_async_to_lds_b128 %0, %1, off offset:16"
               :: "v"(ldsOff), "v"(g) : "memory");
}

// ---------------------------------------------------------------------------
// Batched C[M,N] = A[M,K] * Bt[N,K]^T  (+bias +residual, optional relu),
// bf16 inputs, f32 output (+ optional bf16 shadow copy).
// Global->LDS via async-to-LDS, double buffered. Rows clamped (not guarded):
// clamped rows only feed outputs that are masked at store time.
// ---------------------------------------------------------------------------
__global__ __launch_bounds__(256) void gemm_bt_wmma_kernel(
    const bf16_t* __restrict__ A, long ldA, long sAo, long sAi,
    const bf16_t* __restrict__ Bt, long ldB, long sBo, long sBi,
    float* __restrict__ C, long ldC, long sCo, long sCi,
    const float* __restrict__ bias,
    const float* __restrict__ res, long sRo, long sRi,
    bf16_t* __restrict__ Cbf,
    int M, int N, int K, int innerCnt, int relu)
{
  __shared__ __align__(16) bf16_t lA[2][128 * KP];
  __shared__ __align__(16) bf16_t lB[2][128 * KP];

  int bb = blockIdx.z;
  int bo = bb / innerCnt, bi = bb % innerCnt;
  A  += (long)bo * sAo + (long)bi * sAi;
  Bt += (long)bo * sBo + (long)bi * sBi;
  long cOff = (long)bo * sCo + (long)bi * sCi;
  long rOff = (long)bo * sRo + (long)bi * sRi;

  int m0 = blockIdx.y * 128;
  int n0 = blockIdx.x * 128;
  int t = threadIdx.x;
  int wave = t >> 5, lane = t & 31;
  int wm = (wave & 3) * 32;   // 4 waves along M (2 frag rows each)
  int wn = (wave >> 2) * 64;  // 2 waves along N (4 frag cols each)

  v8f acc[2][4];
  for (int mi = 0; mi < 2; ++mi)
    for (int ni = 0; ni < 4; ++ni)
      for (int e = 0; e < 8; ++e) acc[mi][ni][e] = 0.f;

  int r   = t >> 1;          // 128 rows, 2 threads/row
  int seg = (t & 1) << 4;    // 16-element segment of the 32-wide k tile

  // clamped source rows (safe: they only affect masked outputs)
  int gmc = min(m0 + r, M - 1);
  int gnc = min(n0 + r, N - 1);
  const bf16_t* gA = A  + (long)gmc * ldA + seg;
  const bf16_t* gB = Bt + (long)gnc * ldB + seg;

  unsigned la[2], lb[2];
  la[0] = (unsigned)(uintptr_t)&lA[0][r * KP + seg];
  la[1] = (unsigned)(uintptr_t)&lA[1][r * KP + seg];
  lb[0] = (unsigned)(uintptr_t)&lB[0][r * KP + seg];
  lb[1] = (unsigned)(uintptr_t)&lB[1][r * KP + seg];

  // prologue: prefetch tile 0 into buffer 0  (4 async ops per wave)
  async_cp32(la[0], gA);
  async_cp32(lb[0], gB);

  int nsteps = K >> 5;
  for (int s = 0; s < nsteps; ++s) {
    int cur = s & 1;
    if (s + 1 < nsteps) {
      // prefetch next tile into the other buffer, then wait for current tile
      async_cp32(la[cur ^ 1], gA + (long)(s + 1) * 32);
      async_cp32(lb[cur ^ 1], gB + (long)(s + 1) * 32);
      asm volatile("s_wait_asynccnt 0x4" ::: "memory");
    } else {
      asm volatile("s_wait_asynccnt 0x0" ::: "memory");
    }
    __syncthreads();

    // fragment loads per CDNA5 16-bit A layout:
    // lanes 0-15 hold K = 0..7 (first 16B) and 16..23 (second 16B);
    // lanes 16-31 hold K = 8..15 and 24..31.
    FragU af[2], bfr[4];
    int khalf = (lane >> 4) << 3;
    int lr16  = lane & 15;
    for (int mi = 0; mi < 2; ++mi) {
      int rr = wm + mi * 16 + lr16;
      af[mi].u[0] = *reinterpret_cast<const uint4*>(&lA[cur][rr * KP + khalf]);
      af[mi].u[1] = *reinterpret_cast<const uint4*>(&lA[cur][rr * KP + khalf + 16]);
    }
    for (int ni = 0; ni < 4; ++ni) {
      int rr = wn + ni * 16 + lr16;
      bfr[ni].u[0] = *reinterpret_cast<const uint4*>(&lB[cur][rr * KP + khalf]);
      bfr[ni].u[1] = *reinterpret_cast<const uint4*>(&lB[cur][rr * KP + khalf + 16]);
    }
    for (int mi = 0; mi < 2; ++mi)
      for (int ni = 0; ni < 4; ++ni)
        acc[mi][ni] = __builtin_amdgcn_wmma_f32_16x16x32_bf16(
            false, af[mi].v, false, bfr[ni].v, (short)0, acc[mi][ni], false, false);
    __syncthreads();   // all waves done reading buf[cur] before it is re-filled
  }

  // C/D layout: VGPR e -> M = e + 8*(lane/16), N = lane%16
  int lr = lane & 15, lh = lane >> 4;
  for (int mi = 0; mi < 2; ++mi) {
    for (int ni = 0; ni < 4; ++ni) {
      int gn = n0 + wn + ni * 16 + lr;
      if (gn >= N) continue;
      float bv = bias ? bias[gn] : 0.f;
      int gmBase = m0 + wm + mi * 16 + lh * 8;
      for (int e = 0; e < 8; ++e) {
        int gm = gmBase + e;
        if (gm >= M) continue;
        float v = acc[mi][ni][e] + bv;
        if (res) v += res[rOff + (long)gm * ldC + gn];
        if (relu) v = fmaxf(v, 0.f);
        long ci = cOff + (long)gm * ldC + gn;
        C[ci] = v;
        if (Cbf) Cbf[ci] = f2bf(v);
      }
    }
  }
}

// ---------------------------------------------------------------------------
// Batched tiled transpose + f32->bf16: out[c][r] = bf16(in[r][c])
// ---------------------------------------------------------------------------
__global__ __launch_bounds__(256) void transpose_f32_bf16_kernel(
    const float* __restrict__ in, long ldIn, long inSO, long inSI,
    bf16_t* __restrict__ out, long ldOut, long outSO, long outSI,
    int R, int Cn, int innerCnt)
{
  __shared__ float tile[32][33];
  int bb = blockIdx.z;
  int bo = bb / innerCnt, bi = bb % innerCnt;
  in  += (long)bo * inSO  + (long)bi * inSI;
  out += (long)bo * outSO + (long)bi * outSI;
  int tx = threadIdx.x & 31, ty = threadIdx.x >> 5;
  int c0 = blockIdx.x * 32, r0 = blockIdx.y * 32;
  for (int j = 0; j < 4; ++j) {
    int rr = r0 + ty + j * 8, cc = c0 + tx;
    tile[ty + j * 8][tx] = (rr < R && cc < Cn) ? in[(long)rr * ldIn + cc] : 0.f;
  }
  __syncthreads();
  for (int j = 0; j < 4; ++j) {
    int orow = c0 + ty + j * 8, ocol = r0 + tx;
    if (orow < Cn && ocol < R)
      out[(long)orow * ldOut + ocol] = f2bf(tile[tx][ty + j * 8]);
  }
}

__global__ void cvt_f32_bf16_kernel(const float* __restrict__ in,
                                    bf16_t* __restrict__ out, long n) {
  long i = (long)blockIdx.x * blockDim.x + threadIdx.x;
  if (i < n) out[i] = f2bf(in[i]);
}

// h = emb[X]*sqrt(DM) + posenc ; writes f32 + bf16
__global__ void embed_kernel(const int* __restrict__ X, const float* __restrict__ emb,
                             float* __restrict__ h, bf16_t* __restrict__ hbf,
                             int BS, int S, int DM) {
  long i = (long)blockIdx.x * blockDim.x + threadIdx.x;
  if (i >= (long)BS * DM) return;
  int row = (int)(i / DM), d = (int)(i % DM);
  int s = row % S;
  int tok = X[row];
  float j = (float)(d & ~1);
  float ang = (float)s * __powf(10000.0f, -j / (float)DM);
  float pos = (d & 1) ? __cosf(ang) : __sinf(ang);
  float v = emb[(long)tok * DM + d] * 32.0f + pos;   // sqrt(1024)=32
  h[i] = v;
  hbf[i] = f2bf(v);
}

__device__ __forceinline__ float waveRedSum(float v) {
  for (int o = 16; o > 0; o >>= 1) v += __shfl_xor(v, o, 32);
  return v;
}
__device__ __forceinline__ float waveRedMax(float v) {
  for (int o = 16; o > 0; o >>= 1) v = fmaxf(v, __shfl_xor(v, o, 32));
  return v;
}

// one wave per row of Sk=256; mask + scale + softmax; bf16 output
__global__ __launch_bounds__(256) void softmax_kernel(
    const float* __restrict__ scores, bf16_t* __restrict__ probs,
    const int* __restrict__ dvalid, const int* __restrict__ evalid,
    int mode, int H, int Sq, int Sk, float scale, int nRows)
{
  int wave = threadIdx.x >> 5, lane = threadIdx.x & 31;
  int row = blockIdx.x * 8 + wave;
  if (row >= nRows) return;
  int q  = row % Sq;
  int bh = row / Sq;
  int b  = bh / H;
  int vl = mode ? evalid[b] : min(dvalid[b], q + 1);
  long base = (long)row * Sk;
  float x[8], m = -1e30f;
  for (int i = 0; i < 8; ++i) {
    int col = lane + i * 32;
    float v = scores[base + col] * scale;
    if (col >= vl) v = 1e-6f;
    x[i] = v;
    m = fmaxf(m, v);
  }
  m = waveRedMax(m);
  float s = 0.f;
  for (int i = 0; i < 8; ++i) { x[i] = __expf(x[i] - m); s += x[i]; }
  s = waveRedSum(s);
  float inv = 1.0f / s;
  for (int i = 0; i < 8; ++i) {
    int col = lane + i * 32;
    probs[base + col] = f2bf(x[i] * inv);
  }
}

// one block per row of DM=1024; writes f32 + bf16
__global__ __launch_bounds__(256) void layernorm_kernel(
    const float* __restrict__ x, const float* __restrict__ g, const float* __restrict__ b,
    float* __restrict__ out, bf16_t* __restrict__ outbf, int DM)
{
  __shared__ float red[8];
  __shared__ float mv[2];
  int t = threadIdx.x, wave = t >> 5, lane = t & 31;
  long base = (long)blockIdx.x * DM;
  float v4[4];
  float s = 0.f;
  for (int i = 0; i < 4; ++i) { v4[i] = x[base + t + i * 256]; s += v4[i]; }
  s = waveRedSum(s);
  if (lane == 0) red[wave] = s;
  __syncthreads();
  if (t == 0) { float tot = 0.f; for (int w = 0; w < 8; ++w) tot += red[w]; mv[0] = tot / DM; }
  __syncthreads();
  float m = mv[0];
  float q = 0.f;
  for (int i = 0; i < 4; ++i) { float d = v4[i] - m; q += d * d; }
  q = waveRedSum(q);
  __syncthreads();
  if (lane == 0) red[wave] = q;
  __syncthreads();
  if (t == 0) { float tot = 0.f; for (int w = 0; w < 8; ++w) tot += red[w]; mv[1] = tot / DM; }
  __syncthreads();
  float rstd = rsqrtf(mv[1] + 1e-5f);
  for (int i = 0; i < 4; ++i) {
    int d = t + i * 256;
    float o = (v4[i] - m) * rstd * g[d] + b[d];
    out[base + d] = o;
    outbf[base + d] = f2bf(o);
  }
}

// ---------------------------------------------------------------------------
extern "C" void kernel_launch(void* const* d_in, const int* in_sizes, int n_in,
                              void* d_out, int out_size, void* d_ws, size_t ws_size,
                              hipStream_t stream) {
  const int B = 4, S = 256, DM = 1024, HID = 4096, V = 32000, H = 16, L = 6;
  const int DH = DM / H;          // 64
  const int BS = B * S;           // 1024

  const int*   X         = (const int*)d_in[0];
  const int*   dec_valid = (const int*)d_in[1];
  const float* enc_out   = (const float*)d_in[2];
  const int*   enc_valid = (const int*)d_in[3];
  const float* emb       = (const float*)d_in[4];
  const float* attnW[8]  = { (const float*)d_in[5], (const float*)d_in[6],
                             (const float*)d_in[7], (const float*)d_in[8],
                             (const float*)d_in[9], (const float*)d_in[10],
                             (const float*)d_in[11], (const float*)d_in[12] };
  const float* W1   = (const float*)d_in[13];
  const float* W2   = (const float*)d_in[14];
  const float* Wout = (const float*)d_in[15];
  const float* b1   = (const float*)d_in[16];
  const float* b2   = (const float*)d_in[17];
  const float* bout = (const float*)d_in[18];
  const float* lng[3] = { (const float*)d_in[19], (const float*)d_in[21], (const float*)d_in[23] };
  const float* lnb[3] = { (const float*)d_in[20], (const float*)d_in[22], (const float*)d_in[24] };
  float* out = (float*)d_out;

  char* wsb = (char*)d_ws;
  size_t off = 0;
  auto alloc = [&](size_t bytes) -> void* {
    void* p = wsb + off;
    off = (off + bytes + 255) & ~(size_t)255;
    return p;
  };

  bf16_t* attnT[8];
  for (int i = 0; i < 8; ++i) attnT[i] = (bf16_t*)alloc((size_t)L * DM * DM * 2);
  bf16_t* W1T   = (bf16_t*)alloc((size_t)L * DM * HID * 2);
  bf16_t* W2T   = (bf16_t*)alloc((size_t)L * HID * DM * 2);
  bf16_t* WoutT = (bf16_t*)alloc((size_t)DM * V * 2);
  float*  h     = (float*) alloc((size_t)BS * DM * 4);
  bf16_t* hbf   = (bf16_t*)alloc((size_t)BS * DM * 2);
  bf16_t* encbf = (bf16_t*)alloc((size_t)BS * DM * 2);
  float*  qf    = (float*) alloc((size_t)BS * DM * 4);
  float*  kf    = (float*) alloc((size_t)BS * DM * 4);
  float*  vf    = (float*) alloc((size_t)BS * DM * 4);
  bf16_t* qbf   = (bf16_t*)alloc((size_t)BS * DM * 2);
  bf16_t* kbf   = (bf16_t*)alloc((size_t)BS * DM * 2);
  bf16_t* vt    = (bf16_t*)alloc((size_t)B * H * DH * S * 2);
  float*  scoresf = (float*) alloc((size_t)B * H * S * S * 4);
  bf16_t* probs   = (bf16_t*)alloc((size_t)B * H * S * S * 2);
  float*  attnf   = (float*) alloc((size_t)BS * DM * 4);
  bf16_t* attnbf  = (bf16_t*)alloc((size_t)BS * DM * 2);
  float*  tmp1  = (float*) alloc((size_t)BS * DM * 4);
  float*  yf    = (float*) alloc((size_t)BS * DM * 4);
  bf16_t* ybf   = (bf16_t*)alloc((size_t)BS * DM * 2);
  float*  y2f   = (float*) alloc((size_t)BS * DM * 4);
  bf16_t* y2bf  = (bf16_t*)alloc((size_t)BS * DM * 2);
  float*  a1f   = (float*) alloc((size_t)BS * HID * 4);
  bf16_t* a1bf  = (bf16_t*)alloc((size_t)BS * HID * 2);
  (void)ws_size; (void)n_in; (void)in_sizes; (void)out_size;

  auto gemm = [&](const bf16_t* A, long ldA, long sAo, long sAi,
                  const bf16_t* Bt, long ldB, long sBo, long sBi,
                  float* C, long ldC, long sCo, long sCi,
                  const float* bias, const float* res, long sRo, long sRi,
                  bf16_t* Cbf, int M, int N, int K, int batch, int innerCnt, int relu) {
    dim3 g((unsigned)((N + 127) / 128), (unsigned)((M + 127) / 128), (unsigned)batch);
    gemm_bt_wmma_kernel<<<g, 256, 0, stream>>>(A, ldA, sAo, sAi, Bt, ldB, sBo, sBi,
                                               C, ldC, sCo, sCi, bias, res, sRo, sRi,
                                               Cbf, M, N, K, innerCnt, relu);
  };
  auto transp = [&](const float* in, long ldIn, long inSO, long inSI,
                    bf16_t* o, long ldOut, long outSO, long outSI,
                    int R, int Cn, int batch, int innerCnt) {
    dim3 g((unsigned)((Cn + 31) / 32), (unsigned)((R + 31) / 32), (unsigned)batch);
    transpose_f32_bf16_kernel<<<g, 256, 0, stream>>>(in, ldIn, inSO, inSI,
                                                     o, ldOut, outSO, outSI, R, Cn, innerCnt);
  };

  // ---- weight conversion (fp32 [K,N] -> bf16 [N,K]) ----
  for (int i = 0; i < 8; ++i)
    transp(attnW[i], DM, (long)DM * DM, 0, attnT[i], DM, (long)DM * DM, 0, DM, DM, L, 1);
  transp(W1,   HID, (long)DM * HID, 0, W1T,   DM,  (long)HID * DM, 0, DM,  HID, L, 1);
  transp(W2,   DM,  (long)HID * DM, 0, W2T,   HID, (long)DM * HID, 0, HID, DM,  L, 1);
  transp(Wout, V,   0,              0, WoutT, DM,  0,              0, DM,  V,   1, 1);

  // ---- activations prep ----
  {
    long n = (long)BS * DM;
    cvt_f32_bf16_kernel<<<(unsigned)((n + 255) / 256), 256, 0, stream>>>(enc_out, encbf, n);
    embed_kernel<<<(unsigned)((n + 255) / 256), 256, 0, stream>>>(X, emb, h, hbf, BS, S, DM);
  }

  const long sBH_A = (long)S * DM;     // batch(b) stride in [BS,DM] activations
  const long sS2   = (long)S * S;      // per-head score stride
  const long sVT   = (long)DH * S;     // per-head vt stride

  for (int l = 0; l < L; ++l) {
    size_t wOff  = (size_t)l * DM * DM;
    const float* b1l = b1 + (size_t)l * HID;
    const float* b2l = b2 + (size_t)l * DM;

    // ---------- self attention ----------
    gemm(hbf, DM, 0, 0, attnT[0] + wOff, DM, 0, 0, qf, DM, 0, 0,
         nullptr, nullptr, 0, 0, qbf, BS, DM, DM, 1, 1, 0);
    gemm(hbf, DM, 0, 0, attnT[1] + wOff, DM, 0, 0, kf, DM, 0, 0,
         nullptr, nullptr, 0, 0, kbf, BS, DM, DM, 1, 1, 0);
    gemm(hbf, DM, 0, 0, attnT[2] + wOff, DM, 0, 0, vf, DM, 0, 0,
         nullptr, nullptr, 0, 0, nullptr, BS, DM, DM, 1, 1, 0);
    transp(vf, DM, sBH_A, DH, vt, S, (long)H * sVT, sVT, S, DH, B * H, H);
    gemm(qbf, DM, sBH_A, DH, kbf, DM, sBH_A, DH, scoresf, S, (long)H * sS2, sS2,
         nullptr, nullptr, 0, 0, nullptr, S, S, DH, B * H, H, 0);
    softmax_kernel<<<(unsigned)(B * H * S / 8), 256, 0, stream>>>(
        scoresf, probs, dec_valid, enc_valid, 0, H, S, S, 0.125f, B * H * S);
    gemm(probs, S, (long)H * sS2, sS2, vt, S, (long)H * sVT, sVT,
         attnf, DM, sBH_A, DH, nullptr, nullptr, 0, 0, nullptr, S, DH, S, B * H, H, 0);
    {
      long n = (long)BS * DM;
      cvt_f32_bf16_kernel<<<(unsigned)((n + 255) / 256), 256, 0, stream>>>(attnf, attnbf, n);
    }
    gemm(attnbf, DM, 0, 0, attnT[3] + wOff, DM, 0, 0, tmp1, DM, 0, 0,
         nullptr, h, 0, 0, nullptr, BS, DM, DM, 1, 1, 0);
    layernorm_kernel<<<(unsigned)BS, 256, 0, stream>>>(
        tmp1, lng[0] + (size_t)l * DM, lnb[0] + (size_t)l * DM, yf, ybf, DM);

    // ---------- cross attention ----------
    gemm(ybf,   DM, 0, 0, attnT[4] + wOff, DM, 0, 0, qf, DM, 0, 0,
         nullptr, nullptr, 0, 0, qbf, BS, DM, DM, 1, 1, 0);
    gemm(encbf, DM, 0, 0, attnT[5] + wOff, DM, 0, 0, kf, DM, 0, 0,
         nullptr, nullptr, 0, 0, kbf, BS, DM, DM, 1, 1, 0);
    gemm(encbf, DM, 0, 0, attnT[6] + wOff, DM, 0, 0, vf, DM, 0, 0,
         nullptr, nullptr, 0, 0, nullptr, BS, DM, DM, 1, 1, 0);
    transp(vf, DM, sBH_A, DH, vt, S, (long)H * sVT, sVT, S, DH, B * H, H);
    gemm(qbf, DM, sBH_A, DH, kbf, DM, sBH_A, DH, scoresf, S, (long)H * sS2, sS2,
         nullptr, nullptr, 0, 0, nullptr, S, S, DH, B * H, H, 0);
    softmax_kernel<<<(unsigned)(B * H * S / 8), 256, 0, stream>>>(
        scoresf, probs, dec_valid, enc_valid, 1, H, S, S, 0.125f, B * H * S);
    gemm(probs, S, (long)H * sS2, sS2, vt, S, (long)H * sVT, sVT,
         attnf, DM, sBH_A, DH, nullptr, nullptr, 0, 0, nullptr, S, DH, S, B * H, H, 0);
    {
      long n = (long)BS * DM;
      cvt_f32_bf16_kernel<<<(unsigned)((n + 255) / 256), 256, 0, stream>>>(attnf, attnbf, n);
    }
    gemm(attnbf, DM, 0, 0, attnT[7] + wOff, DM, 0, 0, tmp1, DM, 0, 0,
         nullptr, yf, 0, 0, nullptr, BS, DM, DM, 1, 1, 0);
    layernorm_kernel<<<(unsigned)BS, 256, 0, stream>>>(
        tmp1, lng[1] + (size_t)l * DM, lnb[1] + (size_t)l * DM, y2f, y2bf, DM);

    // ---------- FFN ----------
    gemm(y2bf, DM, 0, 0, W1T + (size_t)l * DM * HID, DM, 0, 0, a1f, HID, 0, 0,
         b1l, nullptr, 0, 0, a1bf, BS, HID, DM, 1, 1, 1);
    gemm(a1bf, HID, 0, 0, W2T + (size_t)l * HID * DM, HID, 0, 0, tmp1, DM, 0, 0,
         b2l, y2f, 0, 0, nullptr, BS, DM, HID, 1, 1, 0);
    layernorm_kernel<<<(unsigned)BS, 256, 0, stream>>>(
        tmp1, lng[2] + (size_t)l * DM, lnb[2] + (size_t)l * DM, h, hbf, DM);
  }

  // ---------- logits ----------
  gemm(hbf, DM, 0, 0, WoutT, DM, 0, 0, out, V, 0, 0,
       bout, nullptr, 0, 0, nullptr, BS, V, DM, 1, 1, 0);
}
